// CAM1_38817914421378
// MI455X (gfx1250) — compile-verified, matrix operand order checked by reading, not run on hardware
//
#include <hip/hip_runtime.h>
#include <hip/hip_bf16.h>

// ---------------------------------------------------------------------------
// Types / helpers
// ---------------------------------------------------------------------------
typedef __attribute__((ext_vector_type(8)))  float  v8f;
typedef __attribute__((ext_vector_type(16))) __bf16 v16bf;

union Frag16 { v16bf v; uint4 u[2]; };

__device__ inline unsigned short f2bf(float x) {
    unsigned int u = __float_as_uint(x);
    unsigned int r = u + 0x7FFFu + ((u >> 16) & 1u);   // round-to-nearest-even
    return (unsigned short)(r >> 16);
}

// Load one 16x32 (A) or 32x16 (B) bf16 WMMA fragment from a buffer laid out
// [row (A) / col (B)][K contiguous].  ISA layout: lanes 0-15 hold K 0..7 and
// 16..23, lanes 16-31 hold K 8..15 and 24..31, row/col = lane%16.
__device__ inline v16bf load_frag(const unsigned short* p, int ld, int lane) {
    const unsigned short* q = p + (lane & 15) * ld + ((lane >> 4) << 3);
    Frag16 f;
    f.u[0] = *(const uint4*)(q);
    f.u[1] = *(const uint4*)(q + 16);
    return f.v;
}

__device__ inline v8f wmma_bf16(v16bf a, v16bf b, v8f c) {
    return __builtin_amdgcn_wmma_f32_16x16x32_bf16(
        /*neg_a=*/false, a, /*neg_b=*/false, b,
        /*c_mod=*/(short)0, c, /*reuse_a=*/false, /*reuse_b=*/false);
}

// Problem constants
#define NB   4
#define CIN  2048
#define HH   64
#define WW   128
#define CR   256
#define CQ   64
#define NTOK 1280          // 10 rows * 128 cols

#define XT_ELEMS (3 * 66 * 32)     // staged x tile elements per K-step
#define XT_REGS  13                // ceil(6336 / 512)

// ---------------------------------------------------------------------------
// Kernel 1a: conv weights f32 [O][C][3][3] -> bf16 [t][O][C] (K contiguous)
// ---------------------------------------------------------------------------
__global__ void k_prep_wconv(const float* __restrict__ w, unsigned short* __restrict__ wb) {
    int i = blockIdx.x * 256 + threadIdx.x;          // i = (t*256 + o)*2048 + c
    if (i >= 9 * CR * CIN) return;
    int c = i & 2047;
    int o = (i >> 11) & 255;
    int t = i >> 19;
    wb[i] = f2bf(w[(o * CIN + c) * 9 + t]);
}

// ---------------------------------------------------------------------------
// Kernel 1b: 1x1 weights f32 [O][C] -> bf16 [O][C]
// ---------------------------------------------------------------------------
__global__ void k_prep_w1(const float* __restrict__ wq, const float* __restrict__ wk,
                          const float* __restrict__ wv,
                          unsigned short* __restrict__ wqb, unsigned short* __restrict__ wkb,
                          unsigned short* __restrict__ wvb) {
    int i = blockIdx.x * 256 + threadIdx.x;
    if (i < CQ * CR) { wqb[i] = f2bf(wq[i]); wkb[i] = f2bf(wk[i]); }
    if (i < CR * CR) { wvb[i] = f2bf(wv[i]); }
}

// ---------------------------------------------------------------------------
// Kernel 2: conv3x3 (implicit GEMM, bf16 WMMA) + BN + ReLU -> r, final
// WG: 512 threads = 16 waves.  Wave w -> 16 output channels; M-tile = 64 cols.
// Register-double-buffered staging: fetch K-step c0+32 while computing c0.
// ---------------------------------------------------------------------------
__global__ __launch_bounds__(512) void k_conv_bn_relu(
    const float* __restrict__ x, const unsigned short* __restrict__ wconv,
    const float* __restrict__ gamma, const float* __restrict__ beta,
    const float* __restrict__ mean, const float* __restrict__ var,
    float* __restrict__ r_out, float* __restrict__ final_out)
{
    __shared__ __align__(16) unsigned short lx[XT_ELEMS];   // [row][col][c] bf16

    const int tid  = threadIdx.x;
    const int lane = tid & 31;
    const int wave = tid >> 5;
    const int wg   = blockIdx.x;               // b*128 + y*2 + xt
    const int x0   = (wg & 1) * 64;
    const int y    = (wg >> 1) & 63;
    const int b    = wg >> 7;
    const int o0   = wave * 16;

    const v8f vz = {0.f, 0.f, 0.f, 0.f, 0.f, 0.f, 0.f, 0.f};
    v8f acc[4] = {vz, vz, vz, vz};

    float sv[XT_REGS];

    // --- prologue: fetch + stage K-step c0 = 0 -----------------------------
    #pragma unroll
    for (int k = 0; k < XT_REGS; ++k) {
        int i = tid + k * 512;
        if (i < XT_ELEMS) {
            int c   = i & 31;
            int rc  = i >> 5;
            int col = rc % 66;
            int row = rc / 66;
            int gy  = y + row - 1;
            int gx  = x0 + col - 1;
            float v = 0.f;
            if ((unsigned)gy < (unsigned)HH && (unsigned)gx < (unsigned)WW)
                v = x[((size_t)(b * CIN + c) * HH + gy) * WW + gx];
            sv[k] = v;
        }
    }
    #pragma unroll
    for (int k = 0; k < XT_REGS; ++k) {
        int i = tid + k * 512;
        if (i < XT_ELEMS) lx[i] = f2bf(sv[k]);
    }
    __syncthreads();

    for (int c0 = 0; c0 < CIN; c0 += 32) {
        const bool more = (c0 + 32) < CIN;

        // fetch next K-step into registers (overlaps with WMMA below)
        if (more) {
            #pragma unroll
            for (int k = 0; k < XT_REGS; ++k) {
                int i = tid + k * 512;
                if (i < XT_ELEMS) {
                    int c   = i & 31;
                    int rc  = i >> 5;
                    int col = rc % 66;
                    int row = rc / 66;
                    int gy  = y + row - 1;
                    int gx  = x0 + col - 1;
                    float v = 0.f;
                    if ((unsigned)gy < (unsigned)HH && (unsigned)gx < (unsigned)WW)
                        v = x[((size_t)(b * CIN + c0 + 32 + c) * HH + gy) * WW + gx];
                    sv[k] = v;
                }
            }
        }

        // compute current K-step: 9 taps x 4 M-subtiles
        #pragma unroll
        for (int t = 0; t < 9; ++t) {
            const int dy = t / 3;
            const int dx = t % 3;
            const unsigned short* wp = wconv + ((size_t)(t * CR + o0)) * CIN + c0;
            v16bf bf = load_frag(wp, CIN, lane);
            if (more)   // warm next K-step's weight lines (global_prefetch_b8)
                __builtin_prefetch(wp + 32 + (size_t)(lane & 15) * CIN, 0, 0);
            v16bf a0 = load_frag(lx + (dy * 66 +  0 + dx) * 32, 32, lane);
            v16bf a1 = load_frag(lx + (dy * 66 + 16 + dx) * 32, 32, lane);
            v16bf a2 = load_frag(lx + (dy * 66 + 32 + dx) * 32, 32, lane);
            v16bf a3 = load_frag(lx + (dy * 66 + 48 + dx) * 32, 32, lane);
            acc[0] = wmma_bf16(a0, bf, acc[0]);
            acc[1] = wmma_bf16(a1, bf, acc[1]);
            acc[2] = wmma_bf16(a2, bf, acc[2]);
            acc[3] = wmma_bf16(a3, bf, acc[3]);
        }
        __syncthreads();            // all waves done reading lx

        if (more) {
            #pragma unroll
            for (int k = 0; k < XT_REGS; ++k) {
                int i = tid + k * 512;
                if (i < XT_ELEMS) lx[i] = f2bf(sv[k]);
            }
        }
        __syncthreads();            // staged data visible for next iteration
    }

    // epilogue: BN + ReLU, write r and final
    const int o   = o0 + (lane & 15);
    const float sc = gamma[o] * rsqrtf(var[o] + 1e-5f);
    const float sh = beta[o] - mean[o] * sc;
    const int mhi = (lane >> 4) << 3;
    #pragma unroll
    for (int ms = 0; ms < 4; ++ms) {
        #pragma unroll
        for (int j = 0; j < 8; ++j) {
            int m = ms * 16 + j + mhi;
            float v = fmaxf(acc[ms][j] * sc + sh, 0.f);
            size_t idx = ((size_t)(b * CR + o) * HH + y) * WW + x0 + m;
            r_out[idx]     = v;
            final_out[idx] = v;
        }
    }
}

// ---------------------------------------------------------------------------
// Kernel 3: q/k 1x1 projections.  WG: 256 thr = 8 waves (4 q + 4 k), M-tile 64.
// Writes f32 pq/pk; for rows 10..19 also bf16 token-major copies qsT/ksT.
// ---------------------------------------------------------------------------
__global__ __launch_bounds__(256) void k_proj_qk(
    const float* __restrict__ r,
    const unsigned short* __restrict__ wq, const unsigned short* __restrict__ wk,
    float* __restrict__ pq, float* __restrict__ pk,
    unsigned short* __restrict__ qsT, unsigned short* __restrict__ ksT)
{
    __shared__ __align__(16) unsigned short lr[64 * 32];   // [col][c]

    const int tid  = threadIdx.x;
    const int lane = tid & 31;
    const int wave = tid >> 5;
    const int wg   = blockIdx.x;
    const int x0   = (wg & 1) * 64;
    const int y    = (wg >> 1) & 63;
    const int b    = wg >> 7;
    const bool isq = (wave < 4);
    const int o0   = (wave & 3) * 16;
    const unsigned short* wptr = (isq ? wq : wk) + (size_t)o0 * CR;

    const v8f vz = {0.f, 0.f, 0.f, 0.f, 0.f, 0.f, 0.f, 0.f};
    v8f acc[4] = {vz, vz, vz, vz};

    for (int c0 = 0; c0 < CR; c0 += 32) {
        for (int i = tid; i < 64 * 32; i += 256) {
            int c = i & 31, col = i >> 5;
            lr[i] = f2bf(r[((size_t)(b * CR + c0 + c) * HH + y) * WW + x0 + col]);
        }
        __syncthreads();
        v16bf bf = load_frag(wptr + c0, CR, lane);
        v16bf a0 = load_frag(lr +  0 * 32, 32, lane);
        v16bf a1 = load_frag(lr + 16 * 32, 32, lane);
        v16bf a2 = load_frag(lr + 32 * 32, 32, lane);
        v16bf a3 = load_frag(lr + 48 * 32, 32, lane);
        acc[0] = wmma_bf16(a0, bf, acc[0]);
        acc[1] = wmma_bf16(a1, bf, acc[1]);
        acc[2] = wmma_bf16(a2, bf, acc[2]);
        acc[3] = wmma_bf16(a3, bf, acc[3]);
        __syncthreads();
    }

    const int o   = o0 + (lane & 15);
    const int mhi = (lane >> 4) << 3;
    float* dst = isq ? pq : pk;
    unsigned short* tdst = isq ? qsT : ksT;
    #pragma unroll
    for (int ms = 0; ms < 4; ++ms) {
        #pragma unroll
        for (int j = 0; j < 8; ++j) {
            int m = ms * 16 + j + mhi;
            float v = acc[ms][j];
            dst[((size_t)(b * CQ + o) * HH + y) * WW + x0 + m] = v;
            if (y >= 10 && y < 20) {
                int tok = (y - 10) * WW + x0 + m;
                tdst[(size_t)(b * NTOK + tok) * CQ + o] = f2bf(v);
            }
        }
    }
}

// ---------------------------------------------------------------------------
// Kernel 4: v 1x1 projection, only rows 10..19.  WG: 512 thr = 16 waves (256 ch).
// Writes f32 vs [B][256][1280] + bf16 copy (channel-major, K=tok contiguous).
// ---------------------------------------------------------------------------
__global__ __launch_bounds__(512) void k_proj_v(
    const float* __restrict__ r, const unsigned short* __restrict__ wv,
    float* __restrict__ vs_out, unsigned short* __restrict__ vs_b)
{
    __shared__ __align__(16) unsigned short lr[64 * 32];

    const int tid  = threadIdx.x;
    const int lane = tid & 31;
    const int wave = tid >> 5;
    const int wg   = blockIdx.x;               // b*20 + yy*2 + xt
    const int x0   = (wg & 1) * 64;
    const int yy   = (wg >> 1) % 10;
    const int y    = 10 + yy;
    const int b    = wg / 20;
    const int o0   = wave * 16;

    const v8f vz = {0.f, 0.f, 0.f, 0.f, 0.f, 0.f, 0.f, 0.f};
    v8f acc[4] = {vz, vz, vz, vz};

    for (int c0 = 0; c0 < CR; c0 += 32) {
        for (int i = tid; i < 64 * 32; i += 512) {
            int c = i & 31, col = i >> 5;
            lr[i] = f2bf(r[((size_t)(b * CR + c0 + c) * HH + y) * WW + x0 + col]);
        }
        __syncthreads();
        v16bf bf = load_frag(wv + (size_t)o0 * CR + c0, CR, lane);
        v16bf a0 = load_frag(lr +  0 * 32, 32, lane);
        v16bf a1 = load_frag(lr + 16 * 32, 32, lane);
        v16bf a2 = load_frag(lr + 32 * 32, 32, lane);
        v16bf a3 = load_frag(lr + 48 * 32, 32, lane);
        acc[0] = wmma_bf16(a0, bf, acc[0]);
        acc[1] = wmma_bf16(a1, bf, acc[1]);
        acc[2] = wmma_bf16(a2, bf, acc[2]);
        acc[3] = wmma_bf16(a3, bf, acc[3]);
        __syncthreads();
    }

    const int o   = o0 + (lane & 15);
    const int mhi = (lane >> 4) << 3;
    #pragma unroll
    for (int ms = 0; ms < 4; ++ms) {
        #pragma unroll
        for (int j = 0; j < 8; ++j) {
            int m = ms * 16 + j + mhi;
            int tok = yy * WW + x0 + m;
            float v = acc[ms][j];
            size_t idx = (size_t)(b * CR + o) * NTOK + tok;
            vs_out[idx] = v;
            vs_b[idx]   = f2bf(v);
        }
    }
}

// ---------------------------------------------------------------------------
// Kernel 5: energy[b,n,m] = sum_c qsT[b,n,c] * ksT[b,m,c].  K = 64.
// WG: 256 thr = 8 waves; wave -> one 16x16 output tile.
// ---------------------------------------------------------------------------
__global__ __launch_bounds__(256) void k_energy(
    const unsigned short* __restrict__ qsT, const unsigned short* __restrict__ ksT,
    float* __restrict__ energy)
{
    const int tid  = threadIdx.x;
    const int lane = tid & 31;
    const int wave = tid >> 5;
    const int wg   = blockIdx.x;               // b*800 + nt*10 + mg
    const int mg   = wg % 10;
    const int nt   = (wg / 10) % 80;
    const int b    = wg / 800;
    const int n0   = nt * 16;
    const int m0   = (mg * 8 + wave) * 16;

    v8f acc = {0.f, 0.f, 0.f, 0.f, 0.f, 0.f, 0.f, 0.f};
    #pragma unroll
    for (int k0 = 0; k0 < CQ; k0 += 32) {
        v16bf a  = load_frag(qsT + (size_t)(b * NTOK + n0) * CQ + k0, CQ, lane);
        v16bf bf = load_frag(ksT + (size_t)(b * NTOK + m0) * CQ + k0, CQ, lane);
        acc = wmma_bf16(a, bf, acc);
    }
    const int m   = m0 + (lane & 15);
    const int mhi = (lane >> 4) << 3;
    #pragma unroll
    for (int j = 0; j < 8; ++j)
        energy[((size_t)b * NTOK + n0 + j + mhi) * NTOK + m] = acc[j];
}

// ---------------------------------------------------------------------------
// Kernel 6: row softmax over 1280 elements; writes bf16 attention weights.
// ---------------------------------------------------------------------------
__global__ __launch_bounds__(256) void k_softmax(
    const float* __restrict__ energy, unsigned short* __restrict__ attn_b)
{
    __shared__ float red[256];
    const int tid = threadIdx.x;
    const size_t row = blockIdx.x;             // b*1280 + n
    const float* e = energy + row * NTOK;

    float vals[5];
    float mx = -3.4e38f;
    #pragma unroll
    for (int i = 0; i < 5; ++i) { vals[i] = e[tid + i * 256]; mx = fmaxf(mx, vals[i]); }
    red[tid] = mx; __syncthreads();
    for (int s = 128; s > 0; s >>= 1) {
        if (tid < s) red[tid] = fmaxf(red[tid], red[tid + s]);
        __syncthreads();
    }
    mx = red[0]; __syncthreads();

    float sum = 0.f;
    #pragma unroll
    for (int i = 0; i < 5; ++i) { vals[i] = __expf(vals[i] - mx); sum += vals[i]; }
    red[tid] = sum; __syncthreads();
    for (int s = 128; s > 0; s >>= 1) {
        if (tid < s) red[tid] += red[tid + s];
        __syncthreads();
    }
    const float inv = 1.f / red[0];

    unsigned short* o = attn_b + row * NTOK;
    #pragma unroll
    for (int i = 0; i < 5; ++i) o[tid + i * 256] = f2bf(vals[i] * inv);
}

// ---------------------------------------------------------------------------
// Kernel 7: out[n][c] = sum_m attn[n][m] * vs[c][m]; final = r + out (rows 10..19).
// WG: 512 thr = 16 waves (256 ch); n-tile 16 per WG; K = 1280.
// ---------------------------------------------------------------------------
__global__ __launch_bounds__(512) void k_out_final(
    const unsigned short* __restrict__ attn_b, const unsigned short* __restrict__ vs_b,
    const float* __restrict__ r, float* __restrict__ final_out)
{
    const int tid  = threadIdx.x;
    const int lane = tid & 31;
    const int wave = tid >> 5;
    const int wg   = blockIdx.x;               // b*80 + nt
    const int nt   = wg % 80;
    const int b    = wg / 80;
    const int n0   = nt * 16;
    const int c0   = wave * 16;

    v8f acc = {0.f, 0.f, 0.f, 0.f, 0.f, 0.f, 0.f, 0.f};
    for (int k0 = 0; k0 < NTOK; k0 += 32) {
        v16bf a  = load_frag(attn_b + (size_t)(b * NTOK + n0) * NTOK + k0, NTOK, lane);
        v16bf bf = load_frag(vs_b   + (size_t)(b * CR   + c0) * NTOK + k0, NTOK, lane);
        acc = wmma_bf16(a, bf, acc);
    }
    const int c   = c0 + (lane & 15);
    const int mhi = (lane >> 4) << 3;
    #pragma unroll
    for (int j = 0; j < 8; ++j) {
        int n  = n0 + j + mhi;
        int y  = 10 + (n >> 7);
        int xc = n & 127;
        size_t idx = ((size_t)(b * CR + c) * HH + y) * WW + xc;
        final_out[idx] = r[idx] + acc[j];
    }
}

// ---------------------------------------------------------------------------
// Host launcher
// ---------------------------------------------------------------------------
extern "C" void kernel_launch(void* const* d_in, const int* in_sizes, int n_in,
                              void* d_out, int out_size, void* d_ws, size_t ws_size,
                              hipStream_t stream) {
    (void)in_sizes; (void)n_in; (void)out_size; (void)ws_size;

    const float* x     = (const float*)d_in[0];
    const float* wconv = (const float*)d_in[1];
    const float* gamma = (const float*)d_in[2];
    const float* beta  = (const float*)d_in[3];
    const float* mean  = (const float*)d_in[4];
    const float* var   = (const float*)d_in[5];
    const float* wq    = (const float*)d_in[6];
    const float* wk    = (const float*)d_in[7];
    const float* wv    = (const float*)d_in[8];

    float* out = (float*)d_out;
    float* final_o  = out;                       // 4*256*64*128
    float* energy_o = out + 8388608;             // 4*1280*1280
    float* pq_o     = out + 14942208;            // 4*64*64*128
    float* pk_o     = out + 17039360;            // 4*64*64*128
    float* vs_o     = out + 19136512;            // 4*256*1280
    float* r_o      = out + 20447232;            // 4*256*64*128

    char* ws = (char*)d_ws;
    unsigned short* wconv_b = (unsigned short*)(ws);                 //  9.4 MB
    unsigned short* wq_b    = (unsigned short*)(ws + 9437184);       //  32 KB
    unsigned short* wk_b    = (unsigned short*)(ws + 9469952);       //  32 KB
    unsigned short* wv_b    = (unsigned short*)(ws + 9502720);       // 128 KB
    unsigned short* qsT     = (unsigned short*)(ws + 9633792);       // 640 KB
    unsigned short* ksT     = (unsigned short*)(ws + 10289152);      // 640 KB
    unsigned short* vs_b    = (unsigned short*)(ws + 10944512);      // 2.5 MB
    unsigned short* attn_b  = (unsigned short*)(ws + 13565952);      // 12.5 MB

    // 1) weight prep
    k_prep_wconv<<<(9 * CR * CIN + 255) / 256, 256, 0, stream>>>(wconv, wconv_b);
    k_prep_w1<<<(CR * CR + 255) / 256, 256, 0, stream>>>(wq, wk, wv, wq_b, wk_b, wv_b);

    // 2) conv3x3 + BN + ReLU -> r, final(=r everywhere for now)
    k_conv_bn_relu<<<NB * HH * (WW / 64), 512, 0, stream>>>(
        x, wconv_b, gamma, beta, mean, var, r_o, final_o);

    // 3) q/k projections (+ bf16 token-major slices for attention)
    k_proj_qk<<<NB * HH * (WW / 64), 256, 0, stream>>>(
        r_o, wq_b, wk_b, pq_o, pk_o, qsT, ksT);

    // 4) v projection on rows 10..19
    k_proj_v<<<NB * 10 * (WW / 64), 512, 0, stream>>>(r_o, wv_b, vs_o, vs_b);

    // 5) energy = qs^T ks
    k_energy<<<NB * 80 * 10, 256, 0, stream>>>(qsT, ksT, energy_o);

    // 6) softmax rows -> bf16 attn
    k_softmax<<<NB * NTOK, 256, 0, stream>>>(energy_o, attn_b);

    // 7) out = vs @ attn^T; final rows 10..19 = r + out (overwrite)
    k_out_final<<<NB * (NTOK / 16), 512, 0, stream>>>(attn_b, vs_b, r_o, final_o);
}
